// TMix_qwen2rwkv_37477884625392
// MI455X (gfx1250) — matrix operand order repaired and checked by reading, not compiled, not run on hardware
//
#include <hip/hip_runtime.h>
#include <hip/hip_bf16.h>
#include <cstdint>
#include <cstddef>

// Problem sizes (fixed by the reference)
#define Bb    2
#define Tt    512
#define Cc    2048
#define Hh    32
#define DhD   64
#define DMIX  32
#define DDEC  64
#define Mrows (Bb * Tt)   // 1024 rows for all GEMMs

typedef __attribute__((ext_vector_type(16))) _Float16 v16h;
typedef __attribute__((ext_vector_type(8)))  _Float16 v8h;
typedef __attribute__((ext_vector_type(8)))  float    v8f;
typedef __attribute__((ext_vector_type(4)))  int      v4i;

// ---------------------------------------------------------------------------
// CDNA5 async global->LDS copy (ASYNCcnt-tracked). Guarded: falls back to a
// plain load/store pair if the builtin is absent on this toolchain.
// Probe-derived prototype: (as1 v4i*, as3 v4i*, imm offset, imm cpol).
// ---------------------------------------------------------------------------
#if __has_builtin(__builtin_amdgcn_global_load_async_to_lds_b128)
#define HAVE_ASYNC_LDS 1
#else
#define HAVE_ASYNC_LDS 0
#endif

__device__ __forceinline__ void wait_async0() {
#if __has_builtin(__builtin_amdgcn_s_wait_asynccnt)
  __builtin_amdgcn_s_wait_asynccnt(0);
#else
  asm volatile("s_wait_asynccnt 0x0" ::: "memory");
#endif
}

__device__ __forceinline__ void cp16_g2l(const _Float16* g, _Float16* l) {
#if HAVE_ASYNC_LDS
  __builtin_amdgcn_global_load_async_to_lds_b128(
      (__attribute__((address_space(1))) v4i*)
          (const __attribute__((address_space(1))) void*)g,
      (__attribute__((address_space(3))) v4i*)
          (__attribute__((address_space(3))) void*)l,
      0, 0);
#else
  *(v8h*)l = *(const v8h*)g;
#endif
}

// ---------------------------------------------------------------------------
// Kernel 0: weight convert + transpose: Wt16[n*K + k] = (f16) W[k*N + n]
// 32x32 tiles through LDS; grid (K/32, N/32).
// ---------------------------------------------------------------------------
__global__ __launch_bounds__(256) void wcvt_kernel(
    const float* __restrict__ W, _Float16* __restrict__ Wt)
{
  __shared__ float sT[32][33];
  const int k0 = blockIdx.x * 32, n0 = blockIdx.y * 32;
  const int tid = threadIdx.x;
  const int cj = tid & 31, ri = tid >> 5;   // ri: 0..7
#pragma unroll
  for (int rr = 0; rr < 32; rr += 8)
    sT[cj][ri + rr] = W[(size_t)(k0 + ri + rr) * Cc + n0 + cj];
  __syncthreads();
#pragma unroll
  for (int rr = 0; rr < 32; rr += 8)
    Wt[(size_t)(n0 + ri + rr) * Cc + k0 + cj] = (_Float16)sT[ri + rr][cj];
}

// ---------------------------------------------------------------------------
// Kernel 1: token shift + mix LoRA + decay LoRA.
// One block per row r = b*T + t. Produces xr/xk/xv (f16) and gh = exp(w_log).
// ---------------------------------------------------------------------------
__global__ __launch_bounds__(256) void prep_kernel(
    const float* __restrict__ x,
    const float* __restrict__ maa_x,  const float* __restrict__ maa_r,
    const float* __restrict__ maa_k,  const float* __restrict__ maa_v,
    const float* __restrict__ maa_w,
    const float* __restrict__ maa_w1, const float* __restrict__ maa_w2,
    const float* __restrict__ tdec,
    const float* __restrict__ td_w1,  const float* __restrict__ td_w2,
    _Float16* __restrict__ xr16, _Float16* __restrict__ xk16,
    _Float16* __restrict__ xv16, float* __restrict__ gh)
{
  __shared__ alignas(16) float sx[Cc];
  __shared__ alignas(16) float sdx[Cc];
  __shared__ alignas(16) float sbuf[Cc];   // xxx, later reused for xw
  __shared__ alignas(16) float smm[4 * DMIX];
  __shared__ alignas(16) float sh[DDEC];
  __shared__ alignas(16) float sred[256];

  const int r   = blockIdx.x;
  const int t   = r % Tt;
  const int tid = threadIdx.x;
  const float* xrow  = x + (size_t)r * Cc;
  const float* xprev = (t == 0) ? nullptr : (xrow - Cc);

  for (int c = tid; c < Cc; c += 256) {
    float xc = xrow[c];
    float xp = xprev ? xprev[c] : 0.0f;
    float dx = xp - xc;
    sx[c]   = xc;
    sdx[c]  = dx;
    sbuf[c] = xc + dx * maa_x[c];
  }
  __syncthreads();

  // mm = tanh(xxx @ maa_w1)  [128] ; maa_w1 is [C, 128] row-major
  {
    const int j    = tid & 127;
    const int half = tid >> 7;
    float acc = 0.0f;
    const int k0 = half * (Cc / 2), k1 = k0 + (Cc / 2);
    for (int k = k0; k < k1; ++k)
      acc = fmaf(sbuf[k], maa_w1[(size_t)k * (4 * DMIX) + j], acc);
    sred[tid] = acc;
    __syncthreads();
    if (tid < 128) smm[tid] = tanhf(sred[tid] + sred[tid + 128]);
    __syncthreads();
  }

  // m[jh][c]; produce xr/xk/xv (f16), xw -> sbuf
  for (int c = tid; c < Cc; c += 256) {
    float m0 = 0.f, m1 = 0.f, m2 = 0.f, m3 = 0.f;
#pragma unroll
    for (int d = 0; d < DMIX; ++d) {
      m0 = fmaf(smm[0 * DMIX + d], maa_w2[(size_t)(0 * DMIX + d) * Cc + c], m0);
      m1 = fmaf(smm[1 * DMIX + d], maa_w2[(size_t)(1 * DMIX + d) * Cc + c], m1);
      m2 = fmaf(smm[2 * DMIX + d], maa_w2[(size_t)(2 * DMIX + d) * Cc + c], m2);
      m3 = fmaf(smm[3 * DMIX + d], maa_w2[(size_t)(3 * DMIX + d) * Cc + c], m3);
    }
    const float xc = sx[c], dxc = sdx[c];
    xr16[(size_t)r * Cc + c] = (_Float16)(xc + dxc * (maa_r[c] + m0));
    xk16[(size_t)r * Cc + c] = (_Float16)(xc + dxc * (maa_k[c] + m1));
    xv16[(size_t)r * Cc + c] = (_Float16)(xc + dxc * (maa_v[c] + m2));
    sbuf[c] = xc + dxc * (maa_w[c] + m3);   // xw
  }
  __syncthreads();

  // h = tanh(xw @ td_w1)  [64] ; td_w1 is [C, 64]
  {
    const int d    = tid & 63;
    const int part = tid >> 6;
    float acc = 0.0f;
    const int k0 = part * (Cc / 4), k1 = k0 + (Cc / 4);
    for (int k = k0; k < k1; ++k)
      acc = fmaf(sbuf[k], td_w1[(size_t)k * DDEC + d], acc);
    sred[tid] = acc;
    __syncthreads();
    if (tid < 64)
      sh[tid] = tanhf(sred[tid] + sred[tid + 64] + sred[tid + 128] + sred[tid + 192]);
    __syncthreads();
  }

  // gh = exp(max(-exp(time_decay + h @ td_w2), -5))
  for (int c = tid; c < Cc; c += 256) {
    float acc = tdec[c];
#pragma unroll
    for (int d = 0; d < DDEC; ++d)
      acc = fmaf(sh[d], td_w2[(size_t)d * Cc + c], acc);
    float wl = -expf(acc);
    wl = fmaxf(wl, -5.0f);
    gh[(size_t)r * Cc + c] = expf(wl);
  }
}

// ---------------------------------------------------------------------------
// Kernel 2: WMMA GEMM.  Out[M,N] = A16[M,K] * Wt16[N,K]^T (+bias) (*(1-gh))
// Block tile 128x128, BK=32, 8 waves (4x2), wave tile 32x64 = 2x4 16x16 frags.
// Double-buffered LDS; tiles staged with async global->LDS b128 copies.
// ---------------------------------------------------------------------------
#define GBM 128
#define GBN 128
#define GBK 32

__device__ __forceinline__ void stage_tile(const _Float16* __restrict__ g,
                                           int gstride, _Float16* l, int tid) {
  // copy 128 rows x 32 halves (8KB): two 16B chunks per thread
#pragma unroll
  for (int cc = 0; cc < 2; ++cc) {
    const int chunk = tid * 2 + cc;        // 0..511
    const int row   = chunk >> 2;          // 0..127
    const int seg   = (chunk & 3) * 8;     // halves
    cp16_g2l(g + (size_t)row * gstride + seg, l + row * GBK + seg);
  }
}

template <bool HAS_BIAS, bool HAS_GH>
__global__ __launch_bounds__(256) void gemm_f16_wmma(
    const _Float16* __restrict__ A,    // [M,K] row-major f16
    const _Float16* __restrict__ Wt,   // [N,K] row-major f16 (transposed W)
    const float*    __restrict__ bias, // [N]
    const float*    __restrict__ ghm,  // [M,N]: out *= (1 - ghm)
    float*          __restrict__ Out,  // [M,N]
    int Kdim, int Ndim)
{
  __shared__ alignas(16) _Float16 sA0[GBM * GBK];
  __shared__ alignas(16) _Float16 sB0[GBN * GBK];
  __shared__ alignas(16) _Float16 sA1[GBM * GBK];
  __shared__ alignas(16) _Float16 sB1[GBN * GBK];

  const int tid  = threadIdx.x;
  const int bm   = blockIdx.x * GBM;
  const int bn   = blockIdx.y * GBN;
  const int wave = tid >> 5;
  const int lane = tid & 31;
  const int wm   = wave >> 1;           // 0..3 -> 32-row slice
  const int wn   = wave & 1;            // 0..1 -> 64-col slice
  const int lm   = lane & 15;
  const int khalf = lane >> 4;

  const _Float16* Abase = A  + (size_t)bm * Kdim;
  const _Float16* Bbase = Wt + (size_t)bn * Kdim;

  v8f acc[2][4];
#pragma unroll
  for (int i = 0; i < 2; ++i)
#pragma unroll
    for (int j = 0; j < 4; ++j)
#pragma unroll
      for (int e = 0; e < 8; ++e) acc[i][j][e] = 0.0f;

  const int nk = Kdim / GBK;            // 64, even

  // prologue: stage tile 0 into buffer 0
  stage_tile(Abase, Kdim, sA0, tid);
  stage_tile(Bbase, Kdim, sB0, tid);
  wait_async0();
  __syncthreads();

  auto step = [&](int kt, _Float16* curA, _Float16* curB,
                  _Float16* nxtA, _Float16* nxtB) {
    if (kt + 1 < nk) {
      stage_tile(Abase + (kt + 1) * GBK, Kdim, nxtA, tid);
      stage_tile(Bbase + (kt + 1) * GBK, Kdim, nxtB, tid);
    }
    // A frag (ISA 16-bit A 16x32 layout): lane m: K 0..7 / 16..23,
    // lane m+16: K 8..15 / 24..31.
    v16h af[2];
#pragma unroll
    for (int fi = 0; fi < 2; ++fi) {
      const int m = wm * 32 + fi * 16 + lm;
      v8h lo = *(const v8h*)(&curA[m * GBK + khalf * 8]);
      v8h hi = *(const v8h*)(&curA[m * GBK + 16 + khalf * 8]);
#pragma unroll
      for (int e = 0; e < 8; ++e) { af[fi][e] = lo[e]; af[fi][8 + e] = hi[e]; }
    }
    // B frag (same K order as A): lane n: K 0..7 / 16..23, lane n+16: K 8..15 / 24..31.
    v16h bf[4];
#pragma unroll
    for (int fj = 0; fj < 4; ++fj) {
      const int n = wn * 64 + fj * 16 + lm;
      v8h lo = *(const v8h*)(&curB[n * GBK + khalf * 8]);
      v8h hi = *(const v8h*)(&curB[n * GBK + 16 + khalf * 8]);
#pragma unroll
      for (int e = 0; e < 8; ++e) { bf[fj][e] = lo[e]; bf[fj][8 + e] = hi[e]; }
    }
#pragma unroll
    for (int fi = 0; fi < 2; ++fi)
#pragma unroll
      for (int fj = 0; fj < 4; ++fj)
        acc[fi][fj] = __builtin_amdgcn_wmma_f32_16x16x32_f16(
            false, af[fi], false, bf[fj], (short)0, acc[fi][fj], false, false);
    if (kt + 1 < nk) wait_async0();
    __syncthreads();
  };

  for (int kt = 0; kt < nk; kt += 2) {
    step(kt,     sA0, sB0, sA1, sB1);
    step(kt + 1, sA1, sB1, sA0, sB0);
  }

  // Epilogue. C/D layout: VGPR e -> row e (lanes 0..15) / row e+8 (16..31).
#pragma unroll
  for (int fj = 0; fj < 4; ++fj) {
    const int col = bn + wn * 64 + fj * 16 + lm;
    const float bv = HAS_BIAS ? bias[col] : 0.0f;
#pragma unroll
    for (int fi = 0; fi < 2; ++fi)
#pragma unroll
      for (int e = 0; e < 8; ++e) {
        const int row = bm + wm * 32 + fi * 16 + e + 8 * khalf;
        const size_t idx = (size_t)row * Ndim + col;
        float v = acc[fi][fj][e] + bv;
        if constexpr (HAS_GH) v *= (1.0f - ghm[idx]);
        Out[idx] = v;
      }
  }
}

// ---------------------------------------------------------------------------
// Kernel 3: GLA scan. One block per (b,h); 64 threads; thread j keeps state
// column S[:,j] (64 f32) in VGPRs; q/k/g broadcast via LDS (ds_load_b128).
// ---------------------------------------------------------------------------
__global__ __launch_bounds__(64) void gla_scan(
    const float* __restrict__ q, const float* __restrict__ k,
    const float* __restrict__ v, const float* __restrict__ gh,
    float* __restrict__ y)
{
  const int bh = blockIdx.x;
  const int b  = bh / Hh;
  const int h  = bh % Hh;
  const int j  = threadIdx.x;

  __shared__ alignas(16) float sq[DhD];
  __shared__ alignas(16) float sk[DhD];
  __shared__ alignas(16) float sg[DhD];

  float S[DhD];
#pragma unroll
  for (int i = 0; i < DhD; ++i) S[i] = 0.0f;

  const float scale = 0.125f;   // Dh^-0.5

  for (int t = 0; t < Tt; ++t) {
    const size_t base = ((size_t)(b * Tt + t)) * Cc + h * DhD;
    sq[j] = q[base + j];
    sk[j] = k[base + j];
    sg[j] = gh[base + j];
    const float vj = v[base + j];
    __syncthreads();

    float acc = 0.0f;
#pragma unroll
    for (int kk = 0; kk < DhD; kk += 4) {
      float4 k4 = *(const float4*)&sk[kk];
      float4 g4 = *(const float4*)&sg[kk];
      float4 q4 = *(const float4*)&sq[kk];
      S[kk + 0] = fmaf(S[kk + 0], g4.x, k4.x * vj); acc = fmaf(q4.x, S[kk + 0], acc);
      S[kk + 1] = fmaf(S[kk + 1], g4.y, k4.y * vj); acc = fmaf(q4.y, S[kk + 1], acc);
      S[kk + 2] = fmaf(S[kk + 2], g4.z, k4.z * vj); acc = fmaf(q4.z, S[kk + 2], acc);
      S[kk + 3] = fmaf(S[kk + 3], g4.w, k4.w * vj); acc = fmaf(q4.w, S[kk + 3], acc);
    }
    y[base + j] = acc * scale;
    __syncthreads();
  }
}

// ---------------------------------------------------------------------------
// Kernel 4: LayerNorm over C, output f16 for the final WMMA GEMM.
// ---------------------------------------------------------------------------
__global__ __launch_bounds__(256) void ln_kernel(
    const float* __restrict__ y, const float* __restrict__ g,
    const float* __restrict__ bta, _Float16* __restrict__ yln)
{
  __shared__ float red[256];
  __shared__ float red2[256];
  const int r = blockIdx.x, tid = threadIdx.x;
  const float* row = y + (size_t)r * Cc;

  float s = 0.f, s2 = 0.f;
  for (int c = tid; c < Cc; c += 256) { float v = row[c]; s += v; s2 = fmaf(v, v, s2); }
  red[tid] = s; red2[tid] = s2;
  __syncthreads();
  for (int off = 128; off > 0; off >>= 1) {
    if (tid < off) { red[tid] += red[tid + off]; red2[tid] += red2[tid + off]; }
    __syncthreads();
  }
  const float mu  = red[0] * (1.0f / Cc);
  const float var = red2[0] * (1.0f / Cc) - mu * mu;
  const float inv = rsqrtf(var + 1e-5f);
  for (int c = tid; c < Cc; c += 256)
    yln[(size_t)r * Cc + c] = (_Float16)((row[c] - mu) * inv * g[c] + bta[c]);
}

// ---------------------------------------------------------------------------
// Host launcher
// ---------------------------------------------------------------------------
extern "C" void kernel_launch(void* const* d_in, const int* in_sizes, int n_in,
                              void* d_out, int out_size, void* d_ws, size_t ws_size,
                              hipStream_t stream) {
  const float* x      = (const float*)d_in[0];
  const float* maa_x  = (const float*)d_in[1];
  const float* maa_r  = (const float*)d_in[2];
  const float* maa_k  = (const float*)d_in[3];
  const float* maa_v  = (const float*)d_in[4];
  const float* maa_w  = (const float*)d_in[5];
  const float* maa_w1 = (const float*)d_in[6];
  const float* maa_w2 = (const float*)d_in[7];
  const float* tdec   = (const float*)d_in[8];
  const float* td_w1  = (const float*)d_in[9];
  const float* td_w2  = (const float*)d_in[10];
  const float* q_w    = (const float*)d_in[11];
  const float* q_b    = (const float*)d_in[12];
  const float* k_w    = (const float*)d_in[13];
  const float* k_b    = (const float*)d_in[14];
  const float* v_w    = (const float*)d_in[15];
  const float* v_b    = (const float*)d_in[16];
  const float* ln_g   = (const float*)d_in[17];
  const float* ln_b   = (const float*)d_in[18];
  const float* o_w    = (const float*)d_in[19];

  // Workspace carving (~88 MB total)
  char* ws = (char*)d_ws;
  size_t off = 0;
  auto carve = [&](size_t bytes) -> void* {
    void* p = ws + off;
    off = (off + bytes + 255) & ~(size_t)255;
    return p;
  };
  const size_t MC = (size_t)Mrows * Cc;
  const size_t CC = (size_t)Cc * Cc;
  _Float16* xr16  = (_Float16*)carve(MC * sizeof(_Float16));
  _Float16* xk16  = (_Float16*)carve(MC * sizeof(_Float16));
  _Float16* xv16  = (_Float16*)carve(MC * sizeof(_Float16));
  float*    gh    = (float*)   carve(MC * sizeof(float));
  float*    qb    = (float*)   carve(MC * sizeof(float));
  float*    kb    = (float*)   carve(MC * sizeof(float));
  float*    vb    = (float*)   carve(MC * sizeof(float));
  float*    yb    = (float*)   carve(MC * sizeof(float));
  _Float16* yln16 = (_Float16*)carve(MC * sizeof(_Float16));
  _Float16* qw16  = (_Float16*)carve(CC * sizeof(_Float16));
  _Float16* kw16  = (_Float16*)carve(CC * sizeof(_Float16));
  _Float16* vw16  = (_Float16*)carve(CC * sizeof(_Float16));
  _Float16* ow16  = (_Float16*)carve(CC * sizeof(_Float16));

  // 0. weight convert + transpose to f16 [N,K]
  dim3 tgrid(Cc / 32, Cc / 32);
  wcvt_kernel<<<tgrid, 256, 0, stream>>>(q_w, qw16);
  wcvt_kernel<<<tgrid, 256, 0, stream>>>(k_w, kw16);
  wcvt_kernel<<<tgrid, 256, 0, stream>>>(v_w, vw16);
  wcvt_kernel<<<tgrid, 256, 0, stream>>>(o_w, ow16);

  // 1. token shift + LoRAs
  prep_kernel<<<Mrows, 256, 0, stream>>>(
      x, maa_x, maa_r, maa_k, maa_v, maa_w, maa_w1, maa_w2,
      tdec, td_w1, td_w2, xr16, xk16, xv16, gh);

  // 2. q/k/v projections (WMMA). k gets the (1-exp(w_log)) epilogue.
  dim3 ggrid(Mrows / GBM, Cc / GBN);
  gemm_f16_wmma<true,  false><<<ggrid, 256, 0, stream>>>(xr16, qw16, q_b, nullptr, qb, Cc, Cc);
  gemm_f16_wmma<true,  true ><<<ggrid, 256, 0, stream>>>(xk16, kw16, k_b, gh,      kb, Cc, Cc);
  gemm_f16_wmma<true,  false><<<ggrid, 256, 0, stream>>>(xv16, vw16, v_b, nullptr, vb, Cc, Cc);

  // 3. GLA recurrence
  gla_scan<<<Bb * Hh, 64, 0, stream>>>(qb, kb, vb, gh, yb);

  // 4. LayerNorm -> f16
  ln_kernel<<<Mrows, 256, 0, stream>>>(yb, ln_g, ln_b, yln16);

  // 5. output projection (WMMA), f32 result straight to d_out
  gemm_f16_wmma<false, false><<<ggrid, 256, 0, stream>>>(yln16, ow16, nullptr, nullptr,
                                                         (float*)d_out, Cc, Cc);
}
